// GroupedQueryAttention_12524124635239
// MI455X (gfx1250) — matrix operand order repaired and checked by reading, not compiled, not run on hardware
//
#include <hip/hip_runtime.h>
#include <hip/hip_bf16.h>
#include <math.h>

typedef __attribute__((ext_vector_type(16))) __bf16 v16bf;
typedef __attribute__((ext_vector_type(8)))  __bf16 v8bf;
typedef __attribute__((ext_vector_type(8)))  float  v8f;
typedef __attribute__((ext_vector_type(4)))  unsigned v4u;
typedef __attribute__((ext_vector_type(8)))  int      v8i;
typedef __attribute__((ext_vector_type(4)))  int      v4i;

#define EMB 4096
#define NH  32
#define NG  8
#define HD  128
#define SEQ 2048
#define GSZ (NH / NG)

#ifndef __has_builtin
#define __has_builtin(x) 0
#endif
#if __has_builtin(__builtin_amdgcn_tensor_load_to_lds)
#define HAVE_TDM 1
#else
#define HAVE_TDM 0
#endif
#if __has_builtin(__builtin_amdgcn_s_wait_tensorcnt)
#define WAIT_TENSOR() __builtin_amdgcn_s_wait_tensorcnt(0)
#else
#define WAIT_TENSOR() asm volatile("s_wait_tensorcnt 0" ::: "memory")
#endif

static __device__ __forceinline__ __bf16 f2bf(float f) {
  union { float f; unsigned u; } x; x.f = f;
  unsigned r = x.u + 0x7fffu + ((x.u >> 16) & 1u);   // round-to-nearest-even
  unsigned short h = (unsigned short)(r >> 16);
  __bf16 b; __builtin_memcpy(&b, &h, 2); return b;
}

// fp32 -> bf16 elementwise (for x)
__global__ __launch_bounds__(256) void convert_bf16_kernel(
    const float* __restrict__ X, __bf16* __restrict__ Xb)
{
  const int i = blockIdx.x * 256 + threadIdx.x;
  Xb[i] = f2bf(X[i]);
}

// W (K x N fp32) -> Wt (N x K bf16): LDS-tiled transpose, coalesced both sides.
__global__ __launch_bounds__(256) void transpose_conv_kernel(
    const float* __restrict__ W, __bf16* __restrict__ Wt, int K, int N)
{
  __shared__ __bf16 tile[32][33];
  const int n0 = blockIdx.x * 32;
  const int k0 = blockIdx.y * 32;
  const int tx = threadIdx.x & 31;
  const int ty = threadIdx.x >> 5;          // 0..7
#pragma unroll
  for (int r = 0; r < 4; ++r)
    tile[ty + 8 * r][tx] = f2bf(W[(size_t)(k0 + ty + 8 * r) * N + n0 + tx]);
  __syncthreads();
#pragma unroll
  for (int r = 0; r < 4; ++r)
    Wt[(size_t)(n0 + ty + 8 * r) * K + k0 + tx] = tile[tx][ty + 8 * r];
}

// ---------------------------------------------------------------------------
// GEMM: C(MxN) = A(MxK,bf16) @ Bt(NxK,bf16)^T + bias, f32 accum/out.
// Block = 256 threads (8 waves), tile 128x128, K-step 32; waves 2(M) x 4(N).
// A tile staged to LDS by the Tensor Data Mover (wave 0 issues one TDM per
// K-step; D# pad feature produces the [32+8]-half padded rows). B-fragments
// are direct contiguous v16bf global loads from the pre-transposed weights
// (B order chosen here, same as A: lane = col, halves = 16 contiguous K from hi*16).
// ---------------------------------------------------------------------------
#define TM 128
#define TN 128
#define TK 32

__global__ __launch_bounds__(256) void gemm_bf16_wmma(
    const __bf16* __restrict__ A, const __bf16* __restrict__ Bt,
    const float* __restrict__ bias, float* __restrict__ C,
    int M, int N, int K)
{
  __shared__ __bf16 As[TM][TK + 8];   // row stride 80B: conflict-free b128 reads

  const int tid  = threadIdx.x;
  const int lane = tid & 31;
  const int wave = tid >> 5;
  const int wm   = wave >> 2;          // 0..1
  const int wn   = wave & 3;           // 0..3
  const int hi   = lane >> 4;
  const int nl   = lane & 15;
  const int bm   = blockIdx.y * TM;
  const int bn   = blockIdx.x * TN;

  v8f zero = {};
  v8f acc[4][2];
#pragma unroll
  for (int i = 0; i < 4; ++i)
#pragma unroll
    for (int j = 0; j < 2; ++j) acc[i][j] = zero;

  for (int k0 = 0; k0 < K; k0 += TK) {
    __syncthreads();                       // previous tile fully consumed
#if HAVE_TDM
    if (wave == 0) {
      const unsigned lds =
          (unsigned)(unsigned long long)(__attribute__((address_space(3))) __bf16*)&As[0][0];
      const unsigned long long ga =
          (unsigned long long)(const void*)(A + (size_t)bm * K + k0);
      // D# group0: count=1 | lds_addr | global_addr[56:0] | type=2
      v4u g0 = { 1u, lds,
                 (unsigned)(ga & 0xffffffffu),
                 (unsigned)((ga >> 32) & 0x01ffffffu) | (2u << 30) };
      // D# group1: data_size=2B, pad_en, pad_interval=16DW, pad_amount=4DW,
      // tensor_dim0=K, tensor_dim1=M, tile_dim0=32, tile_dim1=128,
      // tensor_dim0_stride=K (element units)
      v8i g1 = { (int)((1u << 16) | (1u << 20) | (3u << 22) | (3u << 25)),
                 (int)(((unsigned)K & 0xffffu) << 16),
                 (int)((((unsigned)K >> 16) & 0xffffu) | (((unsigned)M & 0xffffu) << 16)),
                 (int)((((unsigned)M >> 16) & 0xffffu) | (32u << 16)),
                 (int)128,
                 (int)K,
                 0, 0 };
      v4i gz = {};
      v8i gz8 = {};
      __builtin_amdgcn_tensor_load_to_lds(g0, g1, gz, gz, gz8, 0);
      WAIT_TENSOR();
    }
#else
    {
      const int r  = tid >> 1;
      const int ko = (tid & 1) * 16;
      const __bf16* src = A + (size_t)(bm + r) * K + k0 + ko;
      *(v8bf*)&As[r][ko]     = *(const v8bf*)(src);
      *(v8bf*)&As[r][ko + 8] = *(const v8bf*)(src + 8);
    }
#endif
    __syncthreads();                       // tile ready

    // A fragments: lane = row m (nl), halves = K {akb..+15}
    v16bf afr[4];
    const int akb = hi * 16;
#pragma unroll
    for (int mt = 0; mt < 4; ++mt) {
      const __bf16* row = &As[wm * 64 + mt * 16 + nl][0];
#pragma unroll
      for (int j = 0; j < 8; ++j) {
        afr[mt][j]     = row[akb + j];
        afr[mt][8 + j] = row[akb + 8 + j];
      }
    }
    // B fragments: direct 32B contiguous loads from transposed weights
    v16bf bfr[2];
#pragma unroll
    for (int nt = 0; nt < 2; ++nt)
      bfr[nt] = *(const v16bf*)(Bt + (size_t)(bn + wn * 32 + nt * 16 + nl) * K + k0 + hi * 16);

#pragma unroll
    for (int mt = 0; mt < 4; ++mt)
#pragma unroll
      for (int nt = 0; nt < 2; ++nt)
        acc[mt][nt] = __builtin_amdgcn_wmma_f32_16x16x32_bf16(
            false, afr[mt], false, bfr[nt], (short)0, acc[mt][nt], false, false);
  }

#pragma unroll
  for (int mt = 0; mt < 4; ++mt)
#pragma unroll
    for (int nt = 0; nt < 2; ++nt) {
      const int col = bn + wn * 32 + nt * 16 + nl;
      const float bv = bias ? bias[col] : 0.f;
#pragma unroll
      for (int v = 0; v < 8; ++v) {
        const int row = bm + wm * 64 + mt * 16 + v + 8 * hi;
        C[(size_t)row * N + col] = acc[mt][nt][v] + bv;
      }
    }
}

// ---------------------------------------------------------------------------
// Per-head RMSNorm + RoPE; writes bf16 [h][t][d].
// ---------------------------------------------------------------------------
__global__ __launch_bounds__(128) void norm_rope_kernel(
    const float* __restrict__ Y, const float* __restrict__ scale,
    const float* __restrict__ cosb, const float* __restrict__ sinb,
    __bf16* __restrict__ out, int nheads)
{
  const int t = blockIdx.x;
  const int h = blockIdx.y;
  const int d = threadIdx.x;
  const size_t base = ((size_t)t * nheads + h) * HD;
  const float val = Y[base + d];

  __shared__ float red[HD];
  red[d] = val * val;
  __syncthreads();
  for (int s = HD / 2; s > 0; s >>= 1) {
    if (d < s) red[d] += red[d + s];
    __syncthreads();
  }
  const float r  = rsqrtf(red[0] * (1.0f / HD) + 1e-6f);
  const float xn = val * r * scale[d];
  const int  pidx = (d < HD / 2) ? d + HD / 2 : d - HD / 2;
  float pv = Y[base + pidx] * r * scale[pidx];
  pv = (d < HD / 2) ? -pv : pv;
  const float c = cosb[t * HD + d], s_ = sinb[t * HD + d];
  out[((size_t)h * SEQ + t) * HD + d] = f2bf(xn * c + pv * s_);
}

// V: (T, G*D) fp32 -> bf16 transposed [g][d][t] so PV B-fragments are contiguous.
__global__ __launch_bounds__(256) void v_transpose_kernel(
    const float* __restrict__ V, __bf16* __restrict__ out)
{
  const int i = blockIdx.x * 256 + threadIdx.x;   // SEQ * NG*HD elements
  const int t = i >> 10;
  const int c = i & 1023;                          // g*128 + d
  out[(size_t)c * SEQ + t] = f2bf(V[i]);
}

// ---------------------------------------------------------------------------
// Flash attention: block = (head, 128 query rows), 8 waves; each wave owns a
// 16-row query tile and streams causal 32-key blocks. No cross-wave barriers.
// Output written directly as bf16 (A of the final GEMM).
// ---------------------------------------------------------------------------
__global__ __launch_bounds__(256) void attn_kernel(
    const __bf16* __restrict__ Qb, const __bf16* __restrict__ Kb,
    const __bf16* __restrict__ Vt, __bf16* __restrict__ ctx)
{
  __shared__ __bf16 Plds[8][16][32];   // per-wave P transpose staging
  const int tid  = threadIdx.x;
  const int lane = tid & 31;
  const int wave = tid >> 5;
  const int hi   = lane >> 4;
  const int nl   = lane & 15;
  const int h    = blockIdx.x;
  const int g    = h / GSZ;
  const int qbase = blockIdx.y * 128 + wave * 16;
  const int qlast = qbase + 15;
  const float isd = 0.08838834764831845f;   // 1/sqrt(128)

  const __bf16* qh = Qb + ((size_t)h * SEQ + qbase) * HD;
  const __bf16* kh = Kb + (size_t)g * SEQ * HD;
  const __bf16* vh = Vt + (size_t)g * HD * SEQ;

  v16bf qfr[4];
  {
    const __bf16* qrow = qh + (size_t)nl * HD;
    const int kb = hi * 16;
#pragma unroll
    for (int f = 0; f < 4; ++f)
#pragma unroll
      for (int j = 0; j < 8; ++j) {
        qfr[f][j]     = qrow[f * 32 + kb + j];
        qfr[f][8 + j] = qrow[f * 32 + kb + 8 + j];
      }
  }

  v8f zero = {};
  v8f o[8];
#pragma unroll
  for (int nt = 0; nt < 8; ++nt) o[nt] = zero;
  float mrun[8], lrun[8];
#pragma unroll
  for (int v = 0; v < 8; ++v) { mrun[v] = -1e30f; lrun[v] = 0.f; }

  for (int kb0 = 0; kb0 <= qlast; kb0 += 32) {
    v8f s0 = zero, s1 = zero;
#pragma unroll
    for (int f = 0; f < 4; ++f) {
      v16bf b0 = *(const v16bf*)(kh + (size_t)(kb0 + nl) * HD + f * 32 + hi * 16);
      s0 = __builtin_amdgcn_wmma_f32_16x16x32_bf16(false, qfr[f], false, b0, (short)0, s0, false, false);
    }
#pragma unroll
    for (int f = 0; f < 4; ++f) {
      v16bf b1 = *(const v16bf*)(kh + (size_t)(kb0 + 16 + nl) * HD + f * 32 + hi * 16);
      s1 = __builtin_amdgcn_wmma_f32_16x16x32_bf16(false, qfr[f], false, b1, (short)0, s1, false, false);
    }

    const bool partial = (kb0 + 31 > qlast);
    float p0[8], p1[8];
#pragma unroll
    for (int v = 0; v < 8; ++v) {
      float a = s0[v], b = s1[v];
      if (partial) {
        const int qr = qbase + v + 8 * hi;          // C layout: row = v + 8*hi
        if (kb0 + nl > qr)      a = -1e30f;
        if (kb0 + 16 + nl > qr) b = -1e30f;
      }
      p0[v] = a; p1[v] = b;
    }

#pragma unroll
    for (int v = 0; v < 8; ++v) {
      float bm = fmaxf(p0[v], p1[v]);
#pragma unroll
      for (int m = 1; m < 16; m <<= 1) bm = fmaxf(bm, __shfl_xor(bm, m, 32));
      const float mnew  = fmaxf(mrun[v], bm);
      const float alpha = __expf((mrun[v] - mnew) * isd);
      mrun[v] = mnew;
      const float e0 = __expf((p0[v] - mnew) * isd);
      const float e1 = __expf((p1[v] - mnew) * isd);
      float rs = e0 + e1;
#pragma unroll
      for (int m = 1; m < 16; m <<= 1) rs += __shfl_xor(rs, m, 32);
      lrun[v] = lrun[v] * alpha + rs;
#pragma unroll
      for (int nt = 0; nt < 8; ++nt) o[nt][v] *= alpha;
      const int row = v + 8 * hi;
      Plds[wave][row][nl]      = f2bf(e0);
      Plds[wave][row][16 + nl] = f2bf(e1);
    }
    asm volatile("s_wait_dscnt 0" ::: "memory");

    v16bf pf;
    {
      const int kb = hi * 16;
#pragma unroll
      for (int j = 0; j < 8; ++j) {
        pf[j]     = Plds[wave][nl][kb + j];
        pf[8 + j] = Plds[wave][nl][kb + 8 + j];
      }
    }
#pragma unroll
    for (int nt = 0; nt < 8; ++nt) {
      v16bf bv = *(const v16bf*)(vh + (size_t)(nt * 16 + nl) * SEQ + kb0 + hi * 16);
      o[nt] = __builtin_amdgcn_wmma_f32_16x16x32_bf16(false, pf, false, bv, (short)0, o[nt], false, false);
    }
  }

#pragma unroll
  for (int v = 0; v < 8; ++v) {
    const float rinv = 1.0f / lrun[v];
    const int row = qbase + v + 8 * hi;
#pragma unroll
    for (int nt = 0; nt < 8; ++nt)
      ctx[(size_t)row * (NH * HD) + h * HD + nt * 16 + nl] = f2bf(o[nt][v] * rinv);
  }
}

// ---------------------------------------------------------------------------
#define MiB(x) ((size_t)(x) * 1024 * 1024)

extern "C" void kernel_launch(void* const* d_in, const int* in_sizes, int n_in,
                              void* d_out, int out_size, void* d_ws, size_t ws_size,
                              hipStream_t stream)
{
  const float* x      = (const float*)d_in[0];
  // d_in[1] = causal mask (recomputed from indices; unused)
  const float* cosb   = (const float*)d_in[2];
  const float* sinb   = (const float*)d_in[3];
  const float* wq     = (const float*)d_in[4];
  const float* bq     = (const float*)d_in[5];
  const float* wk     = (const float*)d_in[6];
  const float* bk     = (const float*)d_in[7];
  const float* wv     = (const float*)d_in[8];
  const float* bv     = (const float*)d_in[9];
  const float* wo     = (const float*)d_in[10];
  const float* qscale = (const float*)d_in[11];
  const float* kscale = (const float*)d_in[12];
  float* out = (float*)d_out;

  char* ws = (char*)d_ws;
  __bf16* xb   = (__bf16*)(ws);               // 16 MiB  x as bf16
  __bf16* wqT  = (__bf16*)(ws + MiB(16));     // 32 MiB  wq^T bf16 (N x K)
  __bf16* wkT  = (__bf16*)(ws + MiB(48));     //  8 MiB
  __bf16* wvT  = (__bf16*)(ws + MiB(56));     //  8 MiB
  __bf16* woT  = (__bf16*)(ws + MiB(64));     // 32 MiB
  float*  qf   = (float*)(ws + MiB(96));      // 32 MiB  (region reused for ctxb)
  float*  kf   = (float*)(ws + MiB(128));     //  8 MiB
  float*  vf   = (float*)(ws + MiB(136));     //  8 MiB
  __bf16* qb   = (__bf16*)(ws + MiB(144));    // 16 MiB  [h][t][d]
  __bf16* kbt  = (__bf16*)(ws + MiB(160));    //  4 MiB  [g][t][d]
  __bf16* vtb  = (__bf16*)(ws + MiB(164));    //  4 MiB  [g][d][t]
  __bf16* ctxb = (__bf16*)qf;                 // 16 MiB  aliases qf (dead by then)

  dim3 blk(256);
  convert_bf16_kernel<<<dim3((SEQ * EMB) / 256), blk, 0, stream>>>(x, xb);
  transpose_conv_kernel<<<dim3(EMB / 32, EMB / 32), blk, 0, stream>>>(wq, wqT, EMB, EMB);
  transpose_conv_kernel<<<dim3((NG * HD) / 32, EMB / 32), blk, 0, stream>>>(wk, wkT, EMB, NG * HD);
  transpose_conv_kernel<<<dim3((NG * HD) / 32, EMB / 32), blk, 0, stream>>>(wv, wvT, EMB, NG * HD);
  transpose_conv_kernel<<<dim3(EMB / 32, (NH * HD) / 32), blk, 0, stream>>>(wo, woT, NH * HD, EMB);

  gemm_bf16_wmma<<<dim3(EMB / TN, SEQ / TM), blk, 0, stream>>>(xb, wqT, bq, qf, SEQ, EMB, EMB);
  gemm_bf16_wmma<<<dim3((NG * HD) / TN, SEQ / TM), blk, 0, stream>>>(xb, wkT, bk, kf, SEQ, NG * HD, EMB);
  gemm_bf16_wmma<<<dim3((NG * HD) / TN, SEQ / TM), blk, 0, stream>>>(xb, wvT, bv, vf, SEQ, NG * HD, EMB);

  norm_rope_kernel<<<dim3(SEQ, NH), dim3(128), 0, stream>>>(qf, qscale, cosb, sinb, qb, NH);
  norm_rope_kernel<<<dim3(SEQ, NG), dim3(128), 0, stream>>>(kf, kscale, cosb, sinb, kbt, NG);
  v_transpose_kernel<<<dim3((SEQ * NG * HD) / 256), blk, 0, stream>>>(vf, vtb);

  attn_kernel<<<dim3(NH, SEQ / 128), blk, 0, stream>>>(qb, kbt, vtb, ctxb);

  gemm_bf16_wmma<<<dim3(EMB / TN, SEQ / TM), blk, 0, stream>>>(ctxb, woT, nullptr, out, SEQ, EMB, NH * HD);
}
